// MECGraphNet_54391465837043
// MI455X (gfx1250) — compile-verified, matrix operand order checked by reading, not run on hardware
//
#include <hip/hip_runtime.h>

// ---------------------------------------------------------------------------
// MECGraphNet forward for MI455X (gfx1250, wave32, WMMA).
// Dominant cost: per-edge MLP  [E,128]x[128,64] -> relu -> [E,64]x[64,64],
// fused gather->GEMM->scatter-max per 16-edge tile using v_wmma_f32_16x16x32_f16.
// h (12.8MB) and all per-node buffers are L2-resident (192MB L2), so the edge
// kernel is matrix-op bound -> WMMA path.
//
// Round-2 changes (from asm inspection):
//  * GEMM loops are kt-outer / nt-inner with 4 distinct B fragments per kt so
//    the compiler clauses the ds_load_b128s and issues 4 independent WMMAs
//    back-to-back instead of ds_load -> s_wait_dscnt 0 -> wmma serialization.
//  * GEMM2 / gat_xg B fragments are loop-invariant -> hoisted into registers
//    for the whole tile loop (64 VGPRs), removing 8KB/tile of LDS reads.
// ---------------------------------------------------------------------------

typedef __attribute__((ext_vector_type(16))) _Float16 v16h;
typedef __attribute__((ext_vector_type(8)))  _Float16 v8h;
typedef __attribute__((ext_vector_type(8)))  float    v8f;

#define HD   64      // hidden width
#define FNI  16      // node feature in
#define SQ   32      // q outputs
#define LAY  3

__device__ __forceinline__ void atomic_max_f32(float* p, float v) {
  // CDNA5 native float atomic max, no return (STOREcnt tracked).
  asm volatile("global_atomic_max_num_f32 %0, %1, off"
               :: "v"(reinterpret_cast<unsigned long long>(p)), "v"(v)
               : "memory");
}

__device__ __forceinline__ v8f wmma16(v16h a, v16h b, v8f c) {
  // (neg_a, A, neg_b, B, c_mod, C, reuse_a, reuse_b)
  return __builtin_amdgcn_wmma_f32_16x16x32_f16(false, a, false, b, (short)0, c,
                                                false, false);
}

// Load a 16-half fragment chunkwise (two b128 loads): halves [0..7] at p,
// [8..15] at p+16. Matches A/B fragment layout: lanes 0-15 take K {0..7,16..23}
// of a 32-wide k-tile, lanes 16-31 take K {8..15,24..31}.
__device__ __forceinline__ v16h load_frag(const _Float16* p) {
  v8h lo = *(const v8h*)p;
  v8h hi = *(const v8h*)(p + 16);
  v16h r;
#pragma unroll
  for (int i = 0; i < 8; ++i) { r[i] = lo[i]; r[i + 8] = hi[i]; }
  return r;
}

// ---------------------------------------------------------------------------
// Node encoder: h = LN(relu(x @ Wn + bn)) ; also writes f16 shadow for WMMA.
// One wave per node; lane owns features {lane, lane+32}.
// ---------------------------------------------------------------------------
__global__ __launch_bounds__(256)
void encode_kernel(const float* __restrict__ x, const float* __restrict__ Wn,
                   const float* __restrict__ bn, const float* __restrict__ gn,
                   const float* __restrict__ betan,
                   float* __restrict__ h, _Float16* __restrict__ h16, int N) {
  const int node = (blockIdx.x * blockDim.x + threadIdx.x) >> 5;
  const int lane = threadIdx.x & 31;
  if (node >= N) return;
  float a0 = bn[lane], a1 = bn[lane + 32];
#pragma unroll
  for (int k = 0; k < FNI; ++k) {
    const float xv = x[node * FNI + k];
    a0 = __builtin_fmaf(xv, Wn[k * HD + lane], a0);
    a1 = __builtin_fmaf(xv, Wn[k * HD + lane + 32], a1);
  }
  a0 = fmaxf(a0, 0.f);
  a1 = fmaxf(a1, 0.f);
  float s = a0 + a1;
#pragma unroll
  for (int off = 16; off; off >>= 1) s += __shfl_xor(s, off, 32);
  const float mu = s * (1.f / 64.f);
  const float d0 = a0 - mu, d1 = a1 - mu;
  float ss = d0 * d0 + d1 * d1;
#pragma unroll
  for (int off = 16; off; off >>= 1) ss += __shfl_xor(ss, off, 32);
  const float inv = rsqrtf(ss * (1.f / 64.f) + 1e-5f);
  const float o0 = d0 * inv * gn[lane] + betan[lane];
  const float o1 = d1 * inv * gn[lane + 32] + betan[lane + 32];
  h[node * HD + lane] = o0;
  h[node * HD + lane + 32] = o1;
  h16[node * HD + lane] = (_Float16)o0;
  h16[node * HD + lane + 32] = (_Float16)o1;
}

// ---------------------------------------------------------------------------
// Per-layer scratch init: agg=-inf (scatter-max identity), att=0, emax=-inf,
// wsum=0.
// ---------------------------------------------------------------------------
__global__ __launch_bounds__(256)
void init_layer_kernel(float* __restrict__ agg, float* __restrict__ att,
                       float* __restrict__ emax, float* __restrict__ wsum,
                       int N) {
  const int i = blockIdx.x * blockDim.x + threadIdx.x;
  const float ninf = -__builtin_inff();
  if (i < N * HD) { agg[i] = ninf; att[i] = 0.f; }
  if (i < N) { emax[i] = ninf; wsum[i] = 0.f; }
}

// ---------------------------------------------------------------------------
// Fused EdgeConv: per 16-edge tile (one wave):
//   A1 = [h16[dst]|h16[src]]  (16x128 f16, gathered straight into A-fragments)
//   M  = relu(A1 @ Wc1 + bc1) (WMMA, kt-outer/nt-inner, 4 k-steps x 4 n-tiles)
//   R  = M @ Wc2 + bc2        (WMMA, B fragments register-resident)
//   agg[dst] = atomic fmax    (native global_atomic_max_num_f32)
// Weights pre-transposed into LDS in B-fragment-friendly [n][k] order.
// ---------------------------------------------------------------------------
__global__ __launch_bounds__(256)
void edgeconv_kernel(const _Float16* __restrict__ h16,
                     const int* __restrict__ src, const int* __restrict__ dst,
                     const float* __restrict__ Wc1, const float* __restrict__ bc1,
                     const float* __restrict__ Wc2, const float* __restrict__ bc2,
                     float* __restrict__ agg, int E) {
  __shared__ __align__(16) _Float16 wt1[HD * 128];  // wt1[n][k] = Wc1[k][n]
  __shared__ __align__(16) _Float16 wt2[HD * HD];   // wt2[n][k] = Wc2[k][n]
  __shared__ float b1s[HD], b2s[HD];
  __shared__ __align__(16) _Float16 mtile[8][16 * HD];  // per-wave M stage

  const int tid = threadIdx.x;
  for (int i = tid; i < HD * 128; i += 256) {
    const int n = i >> 7, k = i & 127;
    wt1[n * 128 + k] = (_Float16)Wc1[k * HD + n];
  }
  for (int i = tid; i < HD * HD; i += 256) {
    const int n = i >> 6, k = i & 63;
    wt2[n * HD + k] = (_Float16)Wc2[k * HD + n];
  }
  if (tid < HD) { b1s[tid] = bc1[tid]; b2s[tid] = bc2[tid]; }
  __syncthreads();

  const int lane = tid & 31;
  const int wave = tid >> 5;
  const int l15 = lane & 15;
  const int hiSel = lane >> 4;            // 0: K{0..7,16..23}, 1: K{8..15,24..31}
  const int koffA = hiSel * 8;
  _Float16* mt = mtile[wave];

  // GEMM2 B fragments are invariant over the tile loop: keep in registers
  // (8 x v16h = 64 VGPRs) -> no LDS reads in the dependent GEMM2 tail.
  v16h b2r[8];
#pragma unroll
  for (int nt = 0; nt < 4; ++nt)
#pragma unroll
    for (int kt = 0; kt < 2; ++kt)
      b2r[nt * 2 + kt] = load_frag(&wt2[(nt * 16 + l15) * HD + kt * 32 + koffA]);

  const int nTiles = (E + 15) >> 4;
  const int waveG = blockIdx.x * 8 + wave;
  const int nWaves = gridDim.x * 8;

  for (int tile = waveG; tile < nTiles; tile += nWaves) {
    const int ebase = tile * 16;
    const int eRow = min(ebase + l15, E - 1);
    const int nd = dst[eRow];
    const int ns = src[eRow];

    // Prefetch next tile's edge indices (global_prefetch_b8).
    __builtin_prefetch(&dst[min(ebase + nWaves * 16, E - 1)], 0, 1);

    // --- GEMM1 A fragments: row = edge, K 0..63 from h[dst], 64..127 from h[src]
    v16h a1[4];
#pragma unroll
    for (int kt = 0; kt < 4; ++kt) {
      const int node = (kt < 2) ? nd : ns;
      a1[kt] = load_frag(h16 + node * HD + (kt & 1) * 32 + koffA);
    }

    // kt-outer / nt-inner: clause 8 ds_load_b128, then 4 independent WMMAs.
    v8f acc[4] = {{}, {}, {}, {}};
#pragma unroll
    for (int kt = 0; kt < 4; ++kt) {
      v16h b0 = load_frag(&wt1[(0 * 16 + l15) * 128 + kt * 32 + koffA]);
      v16h b1 = load_frag(&wt1[(1 * 16 + l15) * 128 + kt * 32 + koffA]);
      v16h b2 = load_frag(&wt1[(2 * 16 + l15) * 128 + kt * 32 + koffA]);
      v16h b3 = load_frag(&wt1[(3 * 16 + l15) * 128 + kt * 32 + koffA]);
      acc[0] = wmma16(a1[kt], b0, acc[0]);
      acc[1] = wmma16(a1[kt], b1, acc[1]);
      acc[2] = wmma16(a1[kt], b2, acc[2]);
      acc[3] = wmma16(a1[kt], b3, acc[3]);
    }

    // bias + relu, re-stage M row-major as f16 (wave-private LDS slice;
    // DS ops from one wave are in-order, no barrier needed).
#pragma unroll
    for (int nt = 0; nt < 4; ++nt) {
      const int col = nt * 16 + l15;         // C layout: col = lane&15 (+16*nt)
      const float bias = b1s[col];
#pragma unroll
      for (int r = 0; r < 8; ++r) {
        float v = acc[nt][r] + bias;
        v = v > 0.f ? v : 0.f;
        mt[(r + hiSel * 8) * HD + col] = (_Float16)v;
      }
    }

    // --- GEMM2: M(16x64) @ Wc2(64x64), B register-resident.
    v16h a2[2];
#pragma unroll
    for (int kt = 0; kt < 2; ++kt)
      a2[kt] = load_frag(&mt[l15 * HD + kt * 32 + koffA]);

    int drow[8];
#pragma unroll
    for (int r = 0; r < 8; ++r)
      drow[r] = dst[min(ebase + r + hiSel * 8, E - 1)];

    v8f acc2[4] = {{}, {}, {}, {}};
#pragma unroll
    for (int kt = 0; kt < 2; ++kt) {
      acc2[0] = wmma16(a2[kt], b2r[0 * 2 + kt], acc2[0]);
      acc2[1] = wmma16(a2[kt], b2r[1 * 2 + kt], acc2[1]);
      acc2[2] = wmma16(a2[kt], b2r[2 * 2 + kt], acc2[2]);
      acc2[3] = wmma16(a2[kt], b2r[3 * 2 + kt], acc2[3]);
    }

#pragma unroll
    for (int nt = 0; nt < 4; ++nt) {
      const int col = nt * 16 + l15;
      const float bias = b2s[col];
#pragma unroll
      for (int r = 0; r < 8; ++r) {
        const int e = ebase + r + hiSel * 8;
        if (e < E)
          atomic_max_f32(&agg[drow[r] * HD + col], acc2[nt][r] + bias);
      }
    }
  }
}

// ---------------------------------------------------------------------------
// GAT linear: xg = h @ Wg  (N x 64 x 64) via WMMA, 16 nodes per wave.
// All 8 B fragments register-resident across the tile loop.
// ---------------------------------------------------------------------------
__global__ __launch_bounds__(256)
void gat_xg_kernel(const _Float16* __restrict__ h16,
                   const float* __restrict__ Wg, float* __restrict__ xg, int N) {
  __shared__ __align__(16) _Float16 wt[HD * HD];  // wt[n][k] = Wg[k][n]
  const int tid = threadIdx.x;
  for (int i = tid; i < HD * HD; i += 256) {
    const int n = i >> 6, k = i & 63;
    wt[n * HD + k] = (_Float16)Wg[k * HD + n];
  }
  __syncthreads();

  const int lane = tid & 31;
  const int l15 = lane & 15;
  const int hiSel = lane >> 4;
  const int koffA = hiSel * 8;

  v16h br[8];
#pragma unroll
  for (int nt = 0; nt < 4; ++nt)
#pragma unroll
    for (int kt = 0; kt < 2; ++kt)
      br[nt * 2 + kt] = load_frag(&wt[(nt * 16 + l15) * HD + kt * 32 + koffA]);

  const int nTiles = (N + 15) >> 4;
  const int waveG = blockIdx.x * 8 + (tid >> 5);
  const int nWaves = gridDim.x * 8;

  for (int tile = waveG; tile < nTiles; tile += nWaves) {
    const int nbase = tile * 16;
    const int node = min(nbase + l15, N - 1);
    v16h a[2];
#pragma unroll
    for (int kt = 0; kt < 2; ++kt)
      a[kt] = load_frag(h16 + node * HD + kt * 32 + koffA);

    v8f acc[4] = {{}, {}, {}, {}};
#pragma unroll
    for (int kt = 0; kt < 2; ++kt) {
      acc[0] = wmma16(a[kt], br[0 * 2 + kt], acc[0]);
      acc[1] = wmma16(a[kt], br[1 * 2 + kt], acc[1]);
      acc[2] = wmma16(a[kt], br[2 * 2 + kt], acc[2]);
      acc[3] = wmma16(a[kt], br[3 * 2 + kt], acc[3]);
    }
#pragma unroll
    for (int nt = 0; nt < 4; ++nt) {
      const int col = nt * 16 + l15;
#pragma unroll
      for (int r = 0; r < 8; ++r) {
        const int row = nbase + r + hiSel * 8;
        if (row < N) xg[row * HD + col] = acc[nt][r];
      }
    }
  }
}

// a_s = xg . att_src ; a_d = xg . att_dst   (one wave per node)
__global__ __launch_bounds__(256)
void gat_asd_kernel(const float* __restrict__ xg,
                    const float* __restrict__ att_src,
                    const float* __restrict__ att_dst,
                    float* __restrict__ a_s, float* __restrict__ a_d, int N) {
  const int node = (blockIdx.x * blockDim.x + threadIdx.x) >> 5;
  const int lane = threadIdx.x & 31;
  if (node >= N) return;
  const float x0 = xg[node * HD + lane];
  const float x1 = xg[node * HD + lane + 32];
  float s = x0 * att_src[lane] + x1 * att_src[lane + 32];
  float d = x0 * att_dst[lane] + x1 * att_dst[lane + 32];
#pragma unroll
  for (int off = 16; off; off >>= 1) {
    s += __shfl_xor(s, off, 32);
    d += __shfl_xor(d, off, 32);
  }
  if (lane == 0) { a_s[node] = s; a_d[node] = d; }
}

__device__ __forceinline__ void gat_edge(int i, int E, const int* src,
                                         const int* dst, int& s, int& d) {
  if (i < E) { s = src[i]; d = dst[i]; }
  else       { s = d = i - E; }          // GAT self-loops appended
}

// Pass 1: emax[d] = max over incoming edges of leaky_relu(a_s[s]+a_d[d])
__global__ __launch_bounds__(256)
void gat_emax_kernel(const int* __restrict__ src, const int* __restrict__ dst,
                     const float* __restrict__ a_s, const float* __restrict__ a_d,
                     float* __restrict__ emax, int E, int GE) {
  const int i = blockIdx.x * blockDim.x + threadIdx.x;
  if (i >= GE) return;
  int s, d; gat_edge(i, E, src, dst, s, d);
  float e = a_s[s] + a_d[d];
  e = e > 0.f ? e : 0.2f * e;
  atomic_max_f32(&emax[d], e);
}

// Pass 2: wsum[d] += exp(e - emax[d])
__global__ __launch_bounds__(256)
void gat_wsum_kernel(const int* __restrict__ src, const int* __restrict__ dst,
                     const float* __restrict__ a_s, const float* __restrict__ a_d,
                     const float* __restrict__ emax, float* __restrict__ wsum,
                     int E, int GE) {
  const int i = blockIdx.x * blockDim.x + threadIdx.x;
  if (i >= GE) return;
  int s, d; gat_edge(i, E, src, dst, s, d);
  float e = a_s[s] + a_d[d];
  e = e > 0.f ? e : 0.2f * e;
  atomicAdd(&wsum[d], __expf(e - emax[d]));
}

// Pass 3: att[d] += xg[s] * exp(e-emax[d]) / wsum[d]   (one wave per edge,
// lanes cover the 64 features -> coalesced xg reads, spread atomics)
__global__ __launch_bounds__(256)
void gat_att_kernel(const int* __restrict__ src, const int* __restrict__ dst,
                    const float* __restrict__ a_s, const float* __restrict__ a_d,
                    const float* __restrict__ emax, const float* __restrict__ wsum,
                    const float* __restrict__ xg, float* __restrict__ att,
                    int E, int GE) {
  const int lane = threadIdx.x & 31;
  const int waveG = (blockIdx.x * blockDim.x + threadIdx.x) >> 5;
  const int nWaves = (gridDim.x * blockDim.x) >> 5;
  for (int i = waveG; i < GE; i += nWaves) {
    int s, d; gat_edge(i, E, src, dst, s, d);
    float e = a_s[s] + a_d[d];
    e = e > 0.f ? e : 0.2f * e;
    const float coef = __expf(e - emax[d]) / wsum[d];
    atomicAdd(&att[d * HD + lane], xg[s * HD + lane] * coef);
    atomicAdd(&att[d * HD + lane + 32], xg[s * HD + lane + 32] * coef);
  }
}

// h = relu(h + fix(agg) + att + bg); refresh f16 shadow.
__global__ __launch_bounds__(256)
void combine_kernel(float* __restrict__ h, _Float16* __restrict__ h16,
                    const float* __restrict__ agg, const float* __restrict__ att,
                    const float* __restrict__ bg, int total) {
  const int i = blockIdx.x * blockDim.x + threadIdx.x;
  if (i >= total) return;
  const int c = i & (HD - 1);
  float a = agg[i];
  if (a == -__builtin_inff()) a = 0.f;   // PyG: empty scatter-max rows -> 0
  float v = h[i] + a + att[i] + bg[c];
  v = v > 0.f ? v : 0.f;
  h[i] = v;
  h16[i] = (_Float16)v;
}

// ---------------------------------------------------------------------------
// Head: t = h[0]; q = LN(relu(t@Wq1+bq1))@Wq2+bq2 ; v likewise. 64 threads.
// ---------------------------------------------------------------------------
__global__ __launch_bounds__(64)
void head_kernel(const float* __restrict__ h,
                 const float* __restrict__ Wq1, const float* __restrict__ bq1,
                 const float* __restrict__ gq, const float* __restrict__ betaq,
                 const float* __restrict__ Wq2, const float* __restrict__ bq2,
                 const float* __restrict__ Wv1, const float* __restrict__ bv1,
                 const float* __restrict__ gv, const float* __restrict__ betav,
                 const float* __restrict__ Wv2, const float* __restrict__ bv2,
                 float* __restrict__ out) {
  __shared__ float t[HD], tq[HD], tv[HD], stats[4];
  const int c = threadIdx.x;
  t[c] = h[c];
  __syncthreads();
  float aq = bq1[c], av = bv1[c];
#pragma unroll 8
  for (int k = 0; k < HD; ++k) {
    aq = __builtin_fmaf(t[k], Wq1[k * HD + c], aq);
    av = __builtin_fmaf(t[k], Wv1[k * HD + c], av);
  }
  aq = fmaxf(aq, 0.f);
  av = fmaxf(av, 0.f);
  tq[c] = aq; tv[c] = av;
  __syncthreads();
  if (c == 0) {
    float s = 0.f, s2 = 0.f;
    for (int k = 0; k < HD; ++k) s += tq[k];
    const float mu = s * (1.f / 64.f);
    for (int k = 0; k < HD; ++k) { const float d = tq[k] - mu; s2 += d * d; }
    stats[0] = mu; stats[1] = rsqrtf(s2 * (1.f / 64.f) + 1e-5f);
  }
  if (c == 1) {
    float s = 0.f, s2 = 0.f;
    for (int k = 0; k < HD; ++k) s += tv[k];
    const float mu = s * (1.f / 64.f);
    for (int k = 0; k < HD; ++k) { const float d = tv[k] - mu; s2 += d * d; }
    stats[2] = mu; stats[3] = rsqrtf(s2 * (1.f / 64.f) + 1e-5f);
  }
  __syncthreads();
  tq[c] = (aq - stats[0]) * stats[1] * gq[c] + betaq[c];
  tv[c] = (av - stats[2]) * stats[3] * gv[c] + betav[c];
  __syncthreads();
  if (c < SQ) {
    float s = bq2[c];
    for (int k = 0; k < HD; ++k) s = __builtin_fmaf(tq[k], Wq2[k * SQ + c], s);
    out[c] = s;
  }
  if (c == SQ) {
    float s = bv2[0];
    for (int k = 0; k < HD; ++k) s = __builtin_fmaf(tv[k], Wv2[k], s);
    out[SQ] = s;
  }
}

// ---------------------------------------------------------------------------
extern "C" void kernel_launch(void* const* d_in, const int* in_sizes, int n_in,
                              void* d_out, int out_size, void* d_ws,
                              size_t ws_size, hipStream_t stream) {
  const int N = in_sizes[0] / FNI;
  const int E = in_sizes[1] / 2;
  const int GE = E + N;

  const float* x   = (const float*)d_in[0];
  const int*   ei  = (const int*)d_in[1];
  const int*   src = ei;            // PyG row = source
  const int*   dst = ei + E;        // PyG col = target
  const float* Wn = (const float*)d_in[3];
  const float* bn = (const float*)d_in[4];
  const float* gn = (const float*)d_in[5];
  const float* betan = (const float*)d_in[6];
  // d_in[2],[7..10]: edge encoder — computed but unused by the output; skipped.
  const float* Wc1 = (const float*)d_in[11];
  const float* bc1 = (const float*)d_in[12];
  const float* Wc2 = (const float*)d_in[13];
  const float* bc2 = (const float*)d_in[14];
  const float* Wg  = (const float*)d_in[15];
  const float* ats = (const float*)d_in[16];
  const float* atd = (const float*)d_in[17];
  const float* bg  = (const float*)d_in[18];
  const float* Wq1 = (const float*)d_in[19];
  const float* bq1 = (const float*)d_in[20];
  const float* gq  = (const float*)d_in[21];
  const float* betaq = (const float*)d_in[22];
  const float* Wq2 = (const float*)d_in[23];
  const float* bq2 = (const float*)d_in[24];
  const float* Wv1 = (const float*)d_in[25];
  const float* bv1 = (const float*)d_in[26];
  const float* gv  = (const float*)d_in[27];
  const float* betav = (const float*)d_in[28];
  const float* Wv2 = (const float*)d_in[29];
  const float* bv2 = (const float*)d_in[30];

  // Workspace carve-up (~58 MB total).
  float*    h    = (float*)d_ws;                 // N*64 f32
  _Float16* h16  = (_Float16*)(h + (size_t)N * HD);  // N*64 f16
  float*    agg  = (float*)(h16 + (size_t)N * HD);
  float*    att  = agg + (size_t)N * HD;
  float*    xg   = att + (size_t)N * HD;
  float*    a_s  = xg + (size_t)N * HD;
  float*    a_d  = a_s + N;
  float*    emax = a_d + N;
  float*    wsum = emax + N;

  const int waveBlocksN = (N + 7) / 8;                 // 1 wave/node
  const int elemBlocks  = (N * HD + 255) / 256;        // 1 thread/elem
  const int geBlocks    = (GE + 255) / 256;

  encode_kernel<<<waveBlocksN, 256, 0, stream>>>(x, Wn, bn, gn, betan, h, h16, N);

  for (int l = 0; l < LAY; ++l) {
    init_layer_kernel<<<elemBlocks, 256, 0, stream>>>(agg, att, emax, wsum, N);

    edgeconv_kernel<<<640, 256, 0, stream>>>(
        h16, src, dst, Wc1 + (size_t)l * 2 * HD * HD, bc1 + l * HD,
        Wc2 + (size_t)l * HD * HD, bc2 + l * HD, agg, E);

    gat_xg_kernel<<<391, 256, 0, stream>>>(h16, Wg + (size_t)l * HD * HD, xg, N);
    gat_asd_kernel<<<waveBlocksN, 256, 0, stream>>>(xg, ats + l * HD,
                                                    atd + l * HD, a_s, a_d, N);
    gat_emax_kernel<<<geBlocks, 256, 0, stream>>>(src, dst, a_s, a_d, emax, E, GE);
    gat_wsum_kernel<<<geBlocks, 256, 0, stream>>>(src, dst, a_s, a_d, emax, wsum,
                                                  E, GE);
    gat_att_kernel<<<4096, 256, 0, stream>>>(src, dst, a_s, a_d, emax, wsum, xg,
                                             att, E, GE);
    combine_kernel<<<elemBlocks, 256, 0, stream>>>(h, h16, agg, att, bg + l * HD,
                                                   N * HD);
  }

  head_kernel<<<1, 64, 0, stream>>>(h, Wq1, bq1, gq, betaq, Wq2, bq2, Wv1, bv1,
                                    gv, betav, Wv2, bv2, (float*)d_out);
}
